// MRConv_18159121728105
// MI455X (gfx1250) — matrix operand order repaired and checked by reading, not compiled, not run on hardware
//
#include <hip/hip_runtime.h>

// ---- vector types --------------------------------------------------------
typedef __attribute__((ext_vector_type(16))) _Float16 v16h;
typedef __attribute__((ext_vector_type(8)))  _Float16 v8h;
typedef __attribute__((ext_vector_type(4)))  _Float16 v4h;
typedef __attribute__((ext_vector_type(8)))  float    v8f;
typedef __attribute__((ext_vector_type(4)))  float    f32x4;
typedef __attribute__((ext_vector_type(4)))  unsigned u32x4;

// Monotonic float<->u32 order-preserving encoding (for GLOBAL_ATOMIC_MAX_U32)
#define ENC_NEG_INF 0x007FFFFFu   // == enc(-inf)

__device__ __forceinline__ unsigned enc_ord(float f) {
  unsigned u = __float_as_uint(f);
  return (u & 0x80000000u) ? ~u : (u | 0x80000000u);
}
__device__ __forceinline__ float dec_ord_finite(unsigned e) {
  unsigned u = (e & 0x80000000u) ? (e & 0x7FFFFFFFu) : ~e;
  if ((u & 0x7F800000u) == 0x7F800000u) return 0.0f;  // empty segment (-inf) / NaN -> 0
  return __uint_as_float(u);
}

// ---- kernel 1: init scatter-max buffer to enc(-inf) ----------------------
__global__ void mrconv_init(u32x4* __restrict__ xj, int total4) {
  int t = blockIdx.x * blockDim.x + threadIdx.x;
  if (t < total4) {
    u32x4 v = {ENC_NEG_INF, ENC_NEG_INF, ENC_NEG_INF, ENC_NEG_INF};
    xj[t] = v;
  }
}

// ---- kernel 2: per-edge gather-subtract + scatter-max (one wave32/edge) --
__global__ __launch_bounds__(256) void mrconv_edge(
    const float* __restrict__ x, const int* __restrict__ ei,
    unsigned* __restrict__ xj, int E) {
  int e    = (int)((blockIdx.x * blockDim.x + threadIdx.x) >> 5);
  int lane = threadIdx.x & 31;
  if (e >= E) return;
  int s = ei[e];
  int d = ei[E + e];
  int c = lane * 4;                       // 32 lanes x 4 ch = 128 channels
  f32x4 a  = *(const f32x4*)(x + (size_t)s * 128 + c);
  f32x4 bb = *(const f32x4*)(x + (size_t)d * 128 + c);
  f32x4 r  = a - bb;
  unsigned* p = xj + (size_t)d * 128 + c;
#pragma unroll
  for (int j = 0; j < 4; ++j)
    atomicMax(p + j, enc_ord(r[j]));      // -> global_atomic_max_u32 (L2, no return)
}

// ---- kernel 3: build h = [x | finalize(x_j)] in f16 ----------------------
__global__ void mrconv_buildh(const float* __restrict__ x,
                              const unsigned* __restrict__ xj,
                              _Float16* __restrict__ hb, int total4) {
  int t = blockIdx.x * blockDim.x + threadIdx.x;   // total4 = N*64 (4 ch each)
  if (t >= total4) return;
  int m = t >> 6;
  int q = t & 63;
  f32x4 v;
  if (q < 32) {
    v = *(const f32x4*)(x + (size_t)m * 128 + q * 4);
  } else {
    u32x4 e = *(const u32x4*)(xj + (size_t)m * 128 + (q - 32) * 4);
    v[0] = dec_ord_finite(e[0]); v[1] = dec_ord_finite(e[1]);
    v[2] = dec_ord_finite(e[2]); v[3] = dec_ord_finite(e[3]);
  }
  v4h o;
  o[0] = (_Float16)v[0]; o[1] = (_Float16)v[1];
  o[2] = (_Float16)v[2]; o[3] = (_Float16)v[3];
  *(v4h*)(hb + (size_t)m * 256 + q * 4) = o;
}

// ---- kernel 4: W fp32 -> f16 (128 KB, stays L1/L2 resident) --------------
__global__ void mrconv_convw(const float* __restrict__ W,
                             _Float16* __restrict__ Wh, int total4) {
  int t = blockIdx.x * blockDim.x + threadIdx.x;
  if (t >= total4) return;
  f32x4 v = *(const f32x4*)(W + (size_t)t * 4);
  v4h o;
  o[0] = (_Float16)v[0]; o[1] = (_Float16)v[1];
  o[2] = (_Float16)v[2]; o[3] = (_Float16)v[3];
  *(v4h*)(Wh + (size_t)t * 4) = o;
}

// ---- kernel 5: WMMA GEMM  out[N,256] = h[N,256] @ W^T + b, ReLU ----------
// Block = 4 waves; wave w owns N-range [64w, 64w+64), block owns 16 M-rows.
// A frag (16x32 f16): lane(m=lane&15, g=lane>>4) halves = row m at k0+16g and k0+16g+8.
// B frag (32x16 f16): lane(n=lane&15, g) halves = W row n, contiguous 16 halves at k0+16g.
// Software-pipelined: fragments for step k+32 are loaded into fresh registers
// before the 4 WMMAs of step k, so WMMAs only need partial loadcnt waits.
__global__ __launch_bounds__(128) void mrconv_gemm(
    const _Float16* __restrict__ hb, const _Float16* __restrict__ Wh,
    const float* __restrict__ bias, float* __restrict__ out) {
  const int wave = threadIdx.x >> 5;
  const int lane = threadIdx.x & 31;
  const int g    = lane >> 4;
  const int ml   = lane & 15;            // m for A, n for B/C/D
  const int m0   = blockIdx.x * 16;
  const int nb   = wave * 64;

  const _Float16* arow = hb + (size_t)(m0 + ml) * 256 + 16 * g;  // A: k0+16g
  const _Float16* brow = Wh + (size_t)(nb + ml) * 256 + 16 * g;  // B: k0+16g

  v8f acc[4] = {v8f{}, v8f{}, v8f{}, v8f{}};

  // prologue: fragments for k0 = 0
  v16h acur, bcur[4];
  ((v8h*)&acur)[0] = *(const v8h*)(arow);
  ((v8h*)&acur)[1] = *(const v8h*)(arow + 8);
#pragma unroll
  for (int t = 0; t < 4; ++t)
    bcur[t] = *(const v16h*)(brow + (size_t)(16 * t) * 256);

#pragma unroll
  for (int k0 = 0; k0 < 256; k0 += 32) {
    v16h anxt, bnxt[4];
    if (k0 + 32 < 256) {                 // issue next step's loads first
      const int kn = k0 + 32;
      ((v8h*)&anxt)[0] = *(const v8h*)(arow + kn);
      ((v8h*)&anxt)[1] = *(const v8h*)(arow + kn + 8);
#pragma unroll
      for (int t = 0; t < 4; ++t)
        bnxt[t] = *(const v16h*)(brow + (size_t)(16 * t) * 256 + kn);
    }
#pragma unroll
    for (int t = 0; t < 4; ++t)
      acc[t] = __builtin_amdgcn_wmma_f32_16x16x32_f16(
          false, acur, false, bcur[t], (short)0, acc[t], false, false);
    acur = anxt;
#pragma unroll
    for (int t = 0; t < 4; ++t) bcur[t] = bnxt[t];
  }

#pragma unroll
  for (int t = 0; t < 4; ++t) {
    const int n    = nb + 16 * t + ml;
    const float bv = bias[n];
#pragma unroll
    for (int i = 0; i < 8; ++i) {
      const int m = m0 + 8 * g + i;                  // C/D layout: VGPR i -> M = i + 8g
      float v = acc[t][i] + bv;
      out[(size_t)m * 256 + n] = v > 0.0f ? v : 0.0f;
    }
  }
}

// ---- host launcher -------------------------------------------------------
extern "C" void kernel_launch(void* const* d_in, const int* in_sizes, int n_in,
                              void* d_out, int out_size, void* d_ws, size_t ws_size,
                              hipStream_t stream) {
  const float* x  = (const float*)d_in[0];   // [N,128]
  const int*   ei = (const int*)d_in[1];     // [2,E]
  const float* W  = (const float*)d_in[2];   // [256,256] row-major [out,in]
  const float* b  = (const float*)d_in[3];   // [256]
  float* out = (float*)d_out;                // [N,256]

  const int N = in_sizes[0] / 128;
  const int E = in_sizes[1] / 2;

  char* ws = (char*)d_ws;
  unsigned* xj  = (unsigned*)ws;                              // N*128 u32 = 25.6 MB
  size_t xj_bytes = (size_t)N * 128 * sizeof(unsigned);
  _Float16* hbuf = (_Float16*)(ws + xj_bytes);                // N*256 f16 = 25.6 MB
  size_t h_bytes = (size_t)N * 256 * sizeof(_Float16);
  _Float16* Wh   = (_Float16*)(ws + xj_bytes + h_bytes);      // 256*256 f16 = 128 KB

  // 1) init scatter-max buffer to enc(-inf)
  int init4 = (N * 128) / 4;
  mrconv_init<<<(init4 + 255) / 256, 256, 0, stream>>>((u32x4*)xj, init4);

  // 2) edge gather-subtract + atomic scatter-max (one wave per edge)
  mrconv_edge<<<(E + 7) / 8, 256, 0, stream>>>(x, ei, xj, E);

  // 3) finalize + concat -> f16 h buffer
  int h4 = N * 64;
  mrconv_buildh<<<(h4 + 255) / 256, 256, 0, stream>>>(x, xj, hbuf, h4);

  // 4) W -> f16
  int w4 = (256 * 256) / 4;
  mrconv_convw<<<(w4 + 255) / 256, 256, 0, stream>>>(W, Wh, w4);

  // 5) WMMA GEMM + bias + ReLU (N divisible by 16 for N=50000)
  mrconv_gemm<<<N / 16, 128, 0, stream>>>(hbuf, Wh, b, out);
}
